// MoECNBlock_30039001268949
// MI455X (gfx1250) — matrix operand order repaired and checked by reading, not compiled, not run on hardware
//
#include <hip/hip_runtime.h>
#include <hip/hip_bf16.h>

// ---------------- problem constants ----------------
#define N_   32
#define C_   256
#define H_   32
#define W_   32
#define E_   8
#define KTOP 2
#define HID_ 1024
#define T_   (N_ * H_ * W_)        // 32768 tokens
#define BE_  10240                 // int(K*T*CAP/E + 0.5)
#define LN_EPS 1e-6f

typedef __attribute__((ext_vector_type(16))) __bf16 v16bf;
typedef __attribute__((ext_vector_type(8)))  __bf16 bf16x8;
typedef __attribute__((ext_vector_type(8)))  float  v8f;

__device__ __forceinline__ float gelu_exact(float x) {
    return 0.5f * x * (1.0f + erff(x * 0.70710678118654752f));
}

__device__ __forceinline__ v16bf load_frag_split(const __bf16* lo, const __bf16* hi) {
    bf16x8 l = *(const bf16x8*)lo;   // 16B load
    bf16x8 h = *(const bf16x8*)hi;   // 16B load
    v16bf a;
#pragma unroll
    for (int i = 0; i < 8; ++i) { a[i] = l[i]; a[i + 8] = h[i]; }
    return a;
}

// ---------------- K0: zero expert counters ----------------
__global__ void k_zero_counts(int* counts) {
    if (threadIdx.x < E_) counts[threadIdx.x] = 0;
}

// ---------------- K prep: weight transpose + f32->bf16 ----------------
// w1 (E,C,HID) -> w1T (E,HID,C)   [B operand of GEMM1: per-column contiguous K]
__global__ __launch_bounds__(256) void k_cvt_w1(const float* __restrict__ w1, __bf16* __restrict__ w1T) {
    int idx = blockIdx.x * 256 + threadIdx.x;          // E*HID*C = 2^21
    int e   = idx >> 18;                               // HID*C = 262144
    int rem = idx & 262143;
    int n   = rem >> 8;                                // HID index
    int k   = rem & 255;                               // C index
    w1T[idx] = (__bf16)w1[(e * C_ + k) * HID_ + n];
}
// w2 (E,HID,C) -> w2T (E,C,HID)
__global__ __launch_bounds__(256) void k_cvt_w2(const float* __restrict__ w2, __bf16* __restrict__ w2T) {
    int idx = blockIdx.x * 256 + threadIdx.x;
    int e   = idx >> 18;
    int rem = idx & 262143;
    int n   = rem >> 10;                               // C index
    int k   = rem & 1023;                              // HID index
    w2T[idx] = (__bf16)w2[(e * HID_ + k) * C_ + n];
}

// ---------------- K1: depthwise 7x7 conv + bias, NCHW -> NHWC f32 ----------------
__global__ __launch_bounds__(256) void k_dwconv(const float* __restrict__ in,
                                                const float* __restrict__ ker,
                                                const float* __restrict__ bias,
                                                float* __restrict__ xconv) {
    const int nh = blockIdx.x;               // n*H + h
    const int n  = nh >> 5;
    const int h  = nh & 31;
    const int wc = threadIdx.x & 31;         // w coordinate (lane)
    const int c0 = threadIdx.x >> 5;         // 0..7
#pragma unroll 1
    for (int ci = 0; ci < 32; ++ci) {
        const int c = c0 * 32 + ci;
        const float* kp = ker + c * 49;
        float acc = bias[c];
#pragma unroll
        for (int dy = 0; dy < 7; ++dy) {
            const int y = h + dy - 3;
            if ((unsigned)y < (unsigned)H_) {
                const float* ip = in + ((n * C_ + c) * H_ + y) * W_;
#pragma unroll
                for (int dx = 0; dx < 7; ++dx) {
                    const int x = wc + dx - 3;
                    if ((unsigned)x < (unsigned)W_) acc += ip[x] * kp[dy * 7 + dx];
                }
            }
        }
        xconv[(nh * W_ + wc) * C_ + c] = acc;
    }
}

// ---------------- K2: LN + router + top2 softmax + capacity dispatch ----------------
// one wave32 per token; lane l owns channels c = 32*j + l, j=0..7
__global__ __launch_bounds__(256) void k_ln_route(const float* __restrict__ xconv,
                                                  const float* __restrict__ gamma,
                                                  const float* __restrict__ beta,
                                                  const float* __restrict__ rw,   // (C,E)
                                                  __bf16* __restrict__ buf,       // (E*Be, C) bf16
                                                  int* __restrict__ counts,
                                                  int* __restrict__ rowidx,       // (T,2)
                                                  float* __restrict__ gates) {    // (T,2)
    const int lane = threadIdx.x & 31;
    const int wid  = threadIdx.x >> 5;
    const int t    = blockIdx.x * 8 + wid;

    float xv[8];
#pragma unroll
    for (int j = 0; j < 8; ++j) xv[j] = xconv[t * C_ + j * 32 + lane];

    float s = 0.f;
#pragma unroll
    for (int j = 0; j < 8; ++j) s += xv[j];
#pragma unroll
    for (int m = 16; m >= 1; m >>= 1) s += __shfl_xor(s, m);
    const float mu = s * (1.0f / 256.0f);

    float ss = 0.f;
#pragma unroll
    for (int j = 0; j < 8; ++j) { float d = xv[j] - mu; ss += d * d; }
#pragma unroll
    for (int m = 16; m >= 1; m >>= 1) ss += __shfl_xor(ss, m);
    const float inv = 1.0f / sqrtf(ss * (1.0f / 256.0f) + LN_EPS);

    float xn[8];
#pragma unroll
    for (int j = 0; j < 8; ++j) {
        const int c = j * 32 + lane;
        xn[j] = (xv[j] - mu) * inv * gamma[c] + beta[c];
    }

    // router logits
    float lg[8];
#pragma unroll
    for (int e = 0; e < 8; ++e) lg[e] = 0.f;
#pragma unroll
    for (int j = 0; j < 8; ++j) {
        const float v = xn[j];
        const float* rp = rw + (j * 32 + lane) * E_;
#pragma unroll
        for (int e = 0; e < 8; ++e) lg[e] += v * rp[e];
    }
#pragma unroll
    for (int e = 0; e < 8; ++e)
#pragma unroll
        for (int m = 16; m >= 1; m >>= 1) lg[e] += __shfl_xor(lg[e], m);

    // top-2 (strict > matches lax.top_k tie rule), uniform across lanes
    int e0 = 0; float l0 = lg[0];
#pragma unroll
    for (int e = 1; e < 8; ++e) if (lg[e] > l0) { l0 = lg[e]; e0 = e; }
    int e1 = (e0 == 0) ? 1 : 0; float l1 = lg[e1];
#pragma unroll
    for (int e = 0; e < 8; ++e) if (e != e0 && lg[e] > l1) { l1 = lg[e]; e1 = e; }

    const float d  = expf(l1 - l0);
    const float g0 = 1.0f / (1.0f + d);
    const float g1 = d * g0;

    int pos0 = 0, pos1 = 0;
    if (lane == 0) {
        pos0 = atomicAdd(&counts[e0], 1);
        pos1 = atomicAdd(&counts[e1], 1);
    }
    pos0 = __shfl(pos0, 0);
    pos1 = __shfl(pos1, 0);
    const int r0 = (pos0 < BE_) ? e0 * BE_ + pos0 : -1;
    const int r1 = (pos1 < BE_) ? e1 * BE_ + pos1 : -1;

#pragma unroll
    for (int j = 0; j < 8; ++j) {
        const __bf16 b = (__bf16)xn[j];
        const int c = j * 32 + lane;
        if (r0 >= 0) buf[r0 * C_ + c] = b;
        if (r1 >= 0) buf[r1 * C_ + c] = b;
    }
    if (lane == 0) {
        rowidx[2 * t]     = r0;
        rowidx[2 * t + 1] = r1;
        gates[2 * t]      = g0;
        gates[2 * t + 1]  = g1;
    }
}

// ---------------- K3: fused expert MLP via WMMA bf16 ----------------
// Block = 256 thr (8 waves), 64-row tile of one expert.
// GEMM1: h = gelu(A[64x256] @ w1 + b1) computed in 512-col halves into 64KB LDS (bf16)
// GEMM2: y = h[64x1024] @ w2 + b2, f32 accumulators held across halves.
__global__ __launch_bounds__(256) void k_moe_mlp(const __bf16* __restrict__ buf,
                                                 const __bf16* __restrict__ w1T,  // (E,HID,C)
                                                 const __bf16* __restrict__ w2T,  // (E,C,HID)
                                                 const float* __restrict__ b1,    // (E,HID)
                                                 const float* __restrict__ b2,    // (E,C)
                                                 const int* __restrict__ counts,
                                                 float* __restrict__ ybuf) {      // (E*Be, C)
    extern __shared__ char smem[];
    __bf16* hs = (__bf16*)smem;                       // [64][512] bf16 = 64 KB

    const int e  = blockIdx.x / (BE_ / 64);
    const int m0 = (blockIdx.x % (BE_ / 64)) * 64;
    if (m0 >= counts[e]) return;

    const int lane   = threadIdx.x & 31;
    const int wv     = threadIdx.x >> 5;
    const int msub   = wv & 3;                        // 16-row subtile
    const int grp    = wv >> 2;                       // 0/1 N-split
    const int mloc   = lane & 15;
    const int hiHalf = lane >> 4;                     // 0: lanes 0-15, 1: lanes 16-31
    const int koffA  = hiHalf ? 16 : 0;               // A frag K offset
    const int koffB  = hiHalf ? 16 : 0;               // B frag K offset
    const int colB   = lane & 15;

    const int rowA = e * BE_ + m0 + msub * 16 + mloc;
    const __bf16* Abase = buf + rowA * C_;

    v8f acc2[8];
#pragma unroll
    for (int ns = 0; ns < 8; ++ns)
#pragma unroll
        for (int r = 0; r < 8; ++r) acc2[ns][r] = 0.f;

#pragma unroll 1
    for (int half = 0; half < 2; ++half) {
        // ---- GEMM1: A fragments (K = 256 => 8 k-steps of 32) ----
        v16bf a1[8];
#pragma unroll
        for (int ks = 0; ks < 8; ++ks) {
            const __bf16* p = Abase + ks * 32 + koffA;
            a1[ks] = load_frag_split(p, p + 8);
        }
        const int nbaseH = half * 512 + grp * 256;    // global HID column base for this wave
#pragma unroll 1
        for (int ns = 0; ns < 16; ++ns) {
            const int n = nbaseH + ns * 16 + colB;
            v8f acc;
#pragma unroll
            for (int r = 0; r < 8; ++r) acc[r] = 0.f;
            const __bf16* bp = w1T + (e * HID_ + n) * C_ + koffB;
#pragma unroll
            for (int ks = 0; ks < 8; ++ks) {
                v16bf b = *(const v16bf*)(bp + ks * 32);
                acc = __builtin_amdgcn_wmma_f32_16x16x32_bf16(false, a1[ks], false, b,
                                                              (short)0, acc, false, false);
            }
            const float bias = b1[e * HID_ + n];
            const int colLocal = grp * 256 + ns * 16 + colB;   // column within 512-half
#pragma unroll
            for (int r = 0; r < 8; ++r) {
                const float v = gelu_exact(acc[r] + bias);
                const int mrow = msub * 16 + r + (hiHalf ? 8 : 0);
                hs[mrow * 512 + colLocal] = (__bf16)v;
            }
        }
        __syncthreads();

        // ---- GEMM2 partial: K in [half*512, half*512+512) => 16 k-steps ----
#pragma unroll 1
        for (int ks = 0; ks < 16; ++ks) {
            const __bf16* ap = hs + (msub * 16 + mloc) * 512 + ks * 32 + koffA;
            v16bf a = load_frag_split(ap, ap + 8);
            const int kglob = half * 512 + ks * 32;
#pragma unroll
            for (int ns = 0; ns < 8; ++ns) {
                const int n = grp * 128 + ns * 16 + colB;
                v16bf b = *(const v16bf*)(w2T + (e * C_ + n) * HID_ + kglob + koffB);
                acc2[ns] = __builtin_amdgcn_wmma_f32_16x16x32_bf16(false, a, false, b,
                                                                   (short)0, acc2[ns], false, false);
            }
        }
        __syncthreads();
    }

    // ---- epilogue: bias + store y ----
#pragma unroll
    for (int ns = 0; ns < 8; ++ns) {
        const int n = grp * 128 + ns * 16 + colB;
        const float bias = b2[e * C_ + n];
#pragma unroll
        for (int r = 0; r < 8; ++r) {
            const int row = m0 + msub * 16 + r + (hiHalf ? 8 : 0);
            ybuf[(e * BE_ + row) * C_ + n] = acc2[ns][r] + bias;
        }
    }
}

// ---------------- K5: gated combine + layer-scale + residual, NHWC -> NCHW ----------------
__global__ __launch_bounds__(256) void k_combine(const float* __restrict__ in,
                                                 const float* __restrict__ ls,      // (C,1,1)
                                                 const int* __restrict__ rowidx,
                                                 const float* __restrict__ gates,
                                                 const float* __restrict__ ybuf,
                                                 float* __restrict__ out) {
    const int nh = blockIdx.x;           // n*H + h
    const int n  = nh >> 5;
    const int h  = nh & 31;
    const int wc = threadIdx.x & 31;
    const int c0 = threadIdx.x >> 5;
    const int t  = nh * W_ + wc;
    const int r0 = rowidx[2 * t];
    const int r1 = rowidx[2 * t + 1];
    const float g0 = gates[2 * t];
    const float g1 = gates[2 * t + 1];
#pragma unroll 1
    for (int ci = 0; ci < 32; ++ci) {
        const int c = c0 * 32 + ci;
        float a = 0.f;
        if (r0 >= 0) a += g0 * ybuf[r0 * C_ + c];
        if (r1 >= 0) a += g1 * ybuf[r1 * C_ + c];
        const int idx = ((n * C_ + c) * H_ + h) * W_ + wc;
        out[idx] = in[idx] + ls[c] * a;
    }
}

// ---------------- launcher ----------------
extern "C" void kernel_launch(void* const* d_in, const int* in_sizes, int n_in,
                              void* d_out, int out_size, void* d_ws, size_t ws_size,
                              hipStream_t stream) {
    const float* input      = (const float*)d_in[0];
    const float* dw_kernel  = (const float*)d_in[1];
    const float* dw_bias    = (const float*)d_in[2];
    const float* ln_gamma   = (const float*)d_in[3];
    const float* ln_beta    = (const float*)d_in[4];
    const float* router_w   = (const float*)d_in[5];
    const float* w1         = (const float*)d_in[6];
    const float* b1         = (const float*)d_in[7];
    const float* w2         = (const float*)d_in[8];
    const float* b2         = (const float*)d_in[9];
    const float* layer_sc   = (const float*)d_in[10];
    float* out = (float*)d_out;

    char* ws = (char*)d_ws;
    // workspace layout (256B-aligned offsets)
    const size_t oCounts = 0;                                    // 8 ints
    const size_t oRow    = 256;                                  // T*2 ints
    const size_t oGate   = oRow  + (size_t)T_ * 2 * 4;           // T*2 floats
    const size_t oW1T    = oGate + (size_t)T_ * 2 * 4;           // E*HID*C bf16
    const size_t oW2T    = oW1T  + (size_t)E_ * HID_ * C_ * 2;
    const size_t oBuf    = oW2T  + (size_t)E_ * C_ * HID_ * 2;   // E*Be*C bf16
    const size_t oBig    = oBuf  + (size_t)E_ * BE_ * C_ * 2;    // xconv (f32) aliased with ybuf (f32)

    int*    counts = (int*)(ws + oCounts);
    int*    rowidx = (int*)(ws + oRow);
    float*  gatesp = (float*)(ws + oGate);
    __bf16* w1T    = (__bf16*)(ws + oW1T);
    __bf16* w2T    = (__bf16*)(ws + oW2T);
    __bf16* buf    = (__bf16*)(ws + oBuf);
    float*  xconv  = (float*)(ws + oBig);    // lifetime: K1 -> K2
    float*  ybuf   = (float*)(ws + oBig);    // lifetime: K3 -> K5

    k_zero_counts<<<1, 32, 0, stream>>>(counts);
    k_cvt_w1<<<(E_ * HID_ * C_) / 256, 256, 0, stream>>>(w1, w1T);
    k_cvt_w2<<<(E_ * C_ * HID_) / 256, 256, 0, stream>>>(w2, w2T);
    k_dwconv<<<N_ * H_, 256, 0, stream>>>(input, dw_kernel, dw_bias, xconv);
    k_ln_route<<<T_ / 8, 256, 0, stream>>>(xconv, ln_gamma, ln_beta, router_w,
                                           buf, counts, rowidx, gatesp);
    k_moe_mlp<<<E_ * (BE_ / 64), 256, 64 * 512 * 2, stream>>>(buf, w1T, w2T, b1, b2, counts, ybuf);
    k_combine<<<N_ * H_, 256, 0, stream>>>(input, layer_sc, rowidx, gatesp, ybuf, out);
}
